// Prototype_Prompt_Encoder_18287970746953
// MI455X (gfx1250) — matrix operand order repaired and checked by reading, not compiled, not run on hardware
//
#include <hip/hip_runtime.h>

typedef __attribute__((ext_vector_type(2)))  float  v2f;
typedef __attribute__((ext_vector_type(8)))  float  v8f;
typedef __attribute__((ext_vector_type(8)))  __bf16 v8bf;
typedef __attribute__((ext_vector_type(16))) __bf16 v16bf;
typedef __attribute__((ext_vector_type(4)))  unsigned short us4;
typedef unsigned short u16;

#define HW   4096
#define FEAT 256
#define MID  128
#define NTOK 8

// ---------------------------------------------------------------------------
// fp32 -> bf16 hi/lo split (RNE).  x ~= hi + lo, |err| ~ 2^-16 * |x|.
// GEMM uses 3 bf16 WMMA terms: ah*bh + ah*bl + al*bh (al*bl ~2^-16, dropped).
// ---------------------------------------------------------------------------
static __device__ __forceinline__ void split_bf(float x, u16& h, u16& l) {
  unsigned u = __float_as_uint(x);
  unsigned rh = u + 0x7FFFu + ((u >> 16) & 1u);
  h = (u16)(rh >> 16);
  float r = x - __uint_as_float(((unsigned)h) << 16);
  unsigned v = __float_as_uint(r);
  unsigned rl = v + 0x7FFFu + ((v >> 16) & 1u);
  l = (u16)(rl >> 16);
}

// build a 16x32 bf16 A/B fragment from two contiguous 8-element (16B) chunks
static __device__ __forceinline__ v16bf ldfrag(const u16* p0, const u16* p1) {
  v8bf a = *(const v8bf*)p0;
  v8bf b = *(const v8bf*)p1;
  return __builtin_shufflevector(a, b, 0, 1, 2, 3, 4, 5, 6, 7,
                                 8, 9, 10, 11, 12, 13, 14, 15);
}

static __device__ __forceinline__ v8f wmma_bf(v16bf a, v16bf b, v8f c) {
  return __builtin_amdgcn_wmma_f32_16x16x32_bf16(false, a, false, b,
                                                 (short)0, c, false, false);
}

// ---------------------------------------------------------------------------
// Kernel 0: split w_dense1 (128x256) / w_dense2 (256x128) into bf16 hi/lo in
// B-fragment order: idx = (((k>>5)*2 + ((k>>4)&1))*O + o)*16 + (k&15)
// so a lane's fragment (col o, K-block kb, half hl) is 32 contiguous bytes.
// ---------------------------------------------------------------------------
__global__ __launch_bounds__(256)
void pack_kernel(const float* __restrict__ w1, const float* __restrict__ w2,
                 u16* __restrict__ w1ph, u16* __restrict__ w1pl,
                 u16* __restrict__ w2ph, u16* __restrict__ w2pl) {
  int id = blockIdx.x * 256 + threadIdx.x;       // 0..65535
  if (id < 32768) {
    int o = id >> 8, k = id & 255;               // w1: [128 o][256 k]
    u16 h, l; split_bf(w1[o * FEAT + k], h, l);
    int idx = (((k >> 5) * 2 + ((k >> 4) & 1)) * MID + o) * 16 + (k & 15);
    w1ph[idx] = h; w1pl[idx] = l;
  } else {
    int id2 = id - 32768;
    int o = id2 >> 7, k = id2 & 127;             // w2: [256 o][128 k]
    u16 h, l; split_bf(w2[o * MID + k], h, l);
    int idx = (((k >> 5) * 2 + ((k >> 4) & 1)) * FEAT + o) * 16 + (k & 15);
    w2ph[idx] = h; w2pl[idx] = l;
  }
}

// ---------------------------------------------------------------------------
// Kernel 1: sim[b,n] = dot(feat[b,n,:], proto[:])  — one wave per row (fp32)
// ---------------------------------------------------------------------------
__global__ __launch_bounds__(256)
void sim_kernel(const float* __restrict__ feat, const float* __restrict__ proto,
                float* __restrict__ sim) {
  int r = blockIdx.x * 8 + (threadIdx.x >> 5);
  int lane = threadIdx.x & 31;
  const float* fr = feat + (size_t)r * FEAT + lane * 8;
  const float* pr = proto + lane * 8;
  float4 f0 = *(const float4*)fr, f1 = *(const float4*)(fr + 4);
  float4 p0 = *(const float4*)pr, p1 = *(const float4*)(pr + 4);
  float p = f0.x * p0.x + f0.y * p0.y + f0.z * p0.z + f0.w * p0.w +
            f1.x * p1.x + f1.y * p1.y + f1.z * p1.z + f1.w * p1.w;
#pragma unroll
  for (int off = 16; off; off >>= 1) p += __shfl_xor(p, off, 32);
  if (lane == 0) sim[r] = p;
}

// ---------------------------------------------------------------------------
// Kernel 2: dense path. WG = 64 rows of one batch, 8 waves.
//   h   = relu(featk[64x256] @ W1^T + b1)
//   out = (h @ W2^T + b2)  stored transposed out[b][o][n] via LDS staging
// ---------------------------------------------------------------------------
#define XPITCH 264        // bf16 elems/row for feat tile (256 + 8 pad)
#define HPITCH 136        // bf16 elems/row for h tile   (128 + 8 pad)
#define DOUT_OFF 69632    // bytes: union region = max(2*64*264*2, 256*68*4)

__global__ __launch_bounds__(256)
void dense_kernel(const float* __restrict__ feat, const float* __restrict__ sim,
                  const u16* __restrict__ w1ph, const u16* __restrict__ w1pl,
                  const u16* __restrict__ w2ph, const u16* __restrict__ w2pl,
                  const float* __restrict__ bd1, const float* __restrict__ bd2,
                  float* __restrict__ out) {
  __shared__ __align__(16) char smem[DOUT_OFF + 2 * 64 * HPITCH * 2];
  u16*   lds_xh  = (u16*)smem;               // [64][264] hi
  u16*   lds_xl  = lds_xh + 64 * XPITCH;     // [64][264] lo
  float* lds_out = (float*)smem;             // [256][68] (union, after stage A)
  u16*   lds_hh  = (u16*)(smem + DOUT_OFF);  // [64][136] hi
  u16*   lds_hl  = lds_hh + 64 * HPITCH;     // [64][136] lo

  const int b = blockIdx.y;
  const int rowbase = blockIdx.x * 64;
  const int tid = threadIdx.x;
  const int wv = tid >> 5, lane = tid & 31;
  const int hl = lane >> 4, l16 = lane & 15;

  // ---- stage scaled feat tile, split to bf16 hi/lo ----
  const float* fr = feat + ((size_t)b * HW + rowbase) * FEAT;
  const float* sr = sim + b * HW + rowbase;
#pragma unroll
  for (int i = 0; i < 16; ++i) {
    int f4 = tid + 256 * i;
    int row = f4 >> 6, pos = (f4 & 63) << 2;
    float s = 1.0f + sr[row];
    float4 v = *(const float4*)(fr + row * FEAT + pos);
    u16 h0, l0, h1, l1, h2, l2, h3, l3;
    split_bf(v.x * s, h0, l0);
    split_bf(v.y * s, h1, l1);
    split_bf(v.z * s, h2, l2);
    split_bf(v.w * s, h3, l3);
    us4 hh = {h0, h1, h2, h3};
    us4 ll = {l0, l1, l2, l3};
    *(us4*)(lds_xh + row * XPITCH + pos) = hh;
    *(us4*)(lds_xl + row * XPITCH + pos) = ll;
  }
  __syncthreads();

  // ---- Stage A: 64x128 = 2x4 tiles of 32x32, one per wave, K=256 ----
  {
    int mt = wv >> 2, nt = wv & 3;
    int ar0 = mt * 32 + l16;
    int o0 = nt * 32 + l16;
    v8f acc[2][2] = {};
    for (int kb = 0; kb < FEAT; kb += 32) {
      int ka = kb + 16 * hl;
      const u16* pa0h = lds_xh + ar0 * XPITCH + ka;
      const u16* pa0l = lds_xl + ar0 * XPITCH + ka;
      v16bf a0h = ldfrag(pa0h, pa0h + 8);
      v16bf a0l = ldfrag(pa0l, pa0l + 8);
      v16bf a1h = ldfrag(pa0h + 16 * XPITCH, pa0h + 16 * XPITCH + 8);
      v16bf a1l = ldfrag(pa0l + 16 * XPITCH, pa0l + 16 * XPITCH + 8);
      int bb = (((kb >> 5) * 2 + hl) * MID + o0) * 16;
      v16bf b0h = ldfrag(w1ph + bb, w1ph + bb + 8);
      v16bf b0l = ldfrag(w1pl + bb, w1pl + bb + 8);
      v16bf b1h = ldfrag(w1ph + bb + 256, w1ph + bb + 264);
      v16bf b1l = ldfrag(w1pl + bb + 256, w1pl + bb + 264);
      acc[0][0] = wmma_bf(a0h, b0h, acc[0][0]);
      acc[0][0] = wmma_bf(a0h, b0l, acc[0][0]);
      acc[0][0] = wmma_bf(a0l, b0h, acc[0][0]);
      acc[0][1] = wmma_bf(a0h, b1h, acc[0][1]);
      acc[0][1] = wmma_bf(a0h, b1l, acc[0][1]);
      acc[0][1] = wmma_bf(a0l, b1h, acc[0][1]);
      acc[1][0] = wmma_bf(a1h, b0h, acc[1][0]);
      acc[1][0] = wmma_bf(a1h, b0l, acc[1][0]);
      acc[1][0] = wmma_bf(a1l, b0h, acc[1][0]);
      acc[1][1] = wmma_bf(a1h, b1h, acc[1][1]);
      acc[1][1] = wmma_bf(a1h, b1l, acc[1][1]);
      acc[1][1] = wmma_bf(a1l, b1h, acc[1][1]);
    }
#pragma unroll
    for (int ns = 0; ns < 2; ++ns) {
      int o = o0 + ns * 16;
      float bias = bd1[o];
#pragma unroll
      for (int ms = 0; ms < 2; ++ms) {
        int m0 = mt * 32 + ms * 16 + 8 * hl;
        v8f v = acc[ms][ns];
#pragma unroll
        for (int e = 0; e < 8; ++e) {
          float hv = v[e] + bias;
          hv = hv > 0.0f ? hv : 0.0f;
          u16 h, l; split_bf(hv, h, l);
          lds_hh[(m0 + e) * HPITCH + o] = h;
          lds_hl[(m0 + e) * HPITCH + o] = l;
        }
      }
    }
  }
  __syncthreads();

  // ---- Stage B: 64x256 = 2x8 tiles, wave does ntile=wv over both mtiles ----
  {
    int o0 = wv * 32 + l16;
#pragma unroll
    for (int mt = 0; mt < 2; ++mt) {
      v8f acc[2][2] = {};
      int ar = mt * 32 + l16;
      for (int kb = 0; kb < MID; kb += 32) {
        int ka = kb + 16 * hl;
        const u16* pa0h = lds_hh + ar * HPITCH + ka;
        const u16* pa0l = lds_hl + ar * HPITCH + ka;
        v16bf a0h = ldfrag(pa0h, pa0h + 8);
        v16bf a0l = ldfrag(pa0l, pa0l + 8);
        v16bf a1h = ldfrag(pa0h + 16 * HPITCH, pa0h + 16 * HPITCH + 8);
        v16bf a1l = ldfrag(pa0l + 16 * HPITCH, pa0l + 16 * HPITCH + 8);
        int bb = (((kb >> 5) * 2 + hl) * FEAT + o0) * 16;
        v16bf b0h = ldfrag(w2ph + bb, w2ph + bb + 8);
        v16bf b0l = ldfrag(w2pl + bb, w2pl + bb + 8);
        v16bf b1h = ldfrag(w2ph + bb + 256, w2ph + bb + 264);
        v16bf b1l = ldfrag(w2pl + bb + 256, w2pl + bb + 264);
        acc[0][0] = wmma_bf(a0h, b0h, acc[0][0]);
        acc[0][0] = wmma_bf(a0h, b0l, acc[0][0]);
        acc[0][0] = wmma_bf(a0l, b0h, acc[0][0]);
        acc[0][1] = wmma_bf(a0h, b1h, acc[0][1]);
        acc[0][1] = wmma_bf(a0h, b1l, acc[0][1]);
        acc[0][1] = wmma_bf(a0l, b1h, acc[0][1]);
        acc[1][0] = wmma_bf(a1h, b0h, acc[1][0]);
        acc[1][0] = wmma_bf(a1h, b0l, acc[1][0]);
        acc[1][0] = wmma_bf(a1l, b0h, acc[1][0]);
        acc[1][1] = wmma_bf(a1h, b1h, acc[1][1]);
        acc[1][1] = wmma_bf(a1h, b1l, acc[1][1]);
        acc[1][1] = wmma_bf(a1l, b1h, acc[1][1]);
      }
#pragma unroll
      for (int ns = 0; ns < 2; ++ns) {
        int o = o0 + ns * 16;
        float bias = bd2[o];
#pragma unroll
        for (int ms = 0; ms < 2; ++ms) {
          int m0 = mt * 32 + ms * 16 + 8 * hl;
          v8f v = acc[ms][ns];
          float4 lo = make_float4(v[0] + bias, v[1] + bias, v[2] + bias, v[3] + bias);
          float4 hi = make_float4(v[4] + bias, v[5] + bias, v[6] + bias, v[7] + bias);
          *(float4*)(lds_out + o * 68 + m0) = lo;
          *(float4*)(lds_out + o * 68 + m0 + 4) = hi;
        }
      }
    }
  }
  __syncthreads();

  // ---- coalesced transposed flush: out[b][o][rowbase + 0..63] ----
  float* outb = out + (size_t)b * FEAT * HW + rowbase;
#pragma unroll
  for (int i = 0; i < 16; ++i) {
    int idx = tid + 256 * i;
    int o = idx >> 4, q = (idx & 15) << 2;
    float4 v = *(const float4*)(lds_out + o * 68 + q);
    *(float4*)(outb + (size_t)o * HW + q) = v;
  }
}

// ---------------------------------------------------------------------------
// Kernel 3: sparse path. WG = (batch, 128-col half).
//   hs = relu(w_sparse1[128x4096] @ featk[4096xcols] + b_s1)
//   sp = w_sparse2[8x128] @ hs + b_s2 + emb_pos
// ---------------------------------------------------------------------------
#define APITCH 72                 // bf16 elems/row for A chunk (64 + 8 pad)
#define BSTRIDE 24                // bf16 elems per column fragment (16 + 8 pad)
#define BPLANE (128 * BSTRIDE)    // one [kblock][half] plane
#define ABYTES (128 * APITCH * 2) // 18432
#define BBYTES (4 * BPLANE * 2)   // 24576

__global__ __launch_bounds__(256)
void sparse_kernel(const float* __restrict__ feat, const float* __restrict__ sim,
                   const float* __restrict__ ws1, const float* __restrict__ bs1,
                   const float* __restrict__ ws2, const float* __restrict__ bs2,
                   const float* __restrict__ emb_pos, float* __restrict__ outs) {
  __shared__ __align__(16) char smem[2 * ABYTES + 2 * BBYTES];  // 86016 B
  __shared__ float lds_small[128 + NTOK * 128 + NTOK];
  u16*   lds_ah  = (u16*)smem;                        // [128][72] hi
  u16*   lds_al  = lds_ah + 128 * APITCH;             // [128][72] lo
  u16*   lds_bph = (u16*)(smem + 2 * ABYTES);         // [4][128][24] hi
  u16*   lds_bpl = lds_bph + 4 * BPLANE;              // [4][128][24] lo
  float* lds_hs  = (float*)smem;                      // [128][132] (union)

  const int b = blockIdx.y;
  const int cbase = blockIdx.x * 128;
  const int tid = threadIdx.x;
  const int wv = tid >> 5, lane = tid & 31;
  const int hl = lane >> 4, l16 = lane & 15;

  for (int i = tid; i < 128; i += 256) lds_small[i] = bs1[i];
  for (int i = tid; i < NTOK * 128; i += 256) lds_small[128 + i] = ws2[i];
  if (tid < NTOK) lds_small[128 + NTOK * 128 + tid] = bs2[tid];

  const int mt = wv & 3;
  const int ng = wv >> 2;                    // ntiles 2*ng, 2*ng+1
  v8f acc[2][2][2] = {};                     // [ntile][msub][nsub]

  const float* fb = feat + (size_t)b * HW * FEAT + cbase;
  const float* sb = sim + b * HW;

  for (int kb = 0; kb < HW; kb += 64) {
    __syncthreads();
    // ---- A chunk: w_sparse1[m][kb..kb+63], split to bf16 hi/lo ----
    {
      int m = tid >> 4, i4 = (tid & 15) << 2;
#pragma unroll
      for (int r = 0; r < 8; ++r, m += 16) {
        float4 v = *(const float4*)(ws1 + (size_t)m * HW + kb + i4);
        u16 h0, l0, h1, l1, h2, l2, h3, l3;
        split_bf(v.x, h0, l0);
        split_bf(v.y, h1, l1);
        split_bf(v.z, h2, l2);
        split_bf(v.w, h3, l3);
        us4 hh = {h0, h1, h2, h3};
        us4 ll = {l0, l1, l2, l3};
        *(us4*)(lds_ah + m * APITCH + i4) = hh;
        *(us4*)(lds_al + m * APITCH + i4) = ll;
      }
    }
    // ---- B chunk: feat rows kb..kb+63 scaled, split, B-fragment layout ----
    {
      int k = tid >> 5, j4 = (tid & 31) << 2;
#pragma unroll
      for (int r = 0; r < 8; ++r, k += 8) {
        float s = 1.0f + sb[kb + k];
        float4 v = *(const float4*)(fb + (size_t)(kb + k) * FEAT + j4);
        int plane = ((k >> 5) * 2 + ((k >> 4) & 1)) * BPLANE;
        int e = k & 15;
        u16 h, l;
        split_bf(v.x * s, h, l);
        lds_bph[plane + (j4 + 0) * BSTRIDE + e] = h;
        lds_bpl[plane + (j4 + 0) * BSTRIDE + e] = l;
        split_bf(v.y * s, h, l);
        lds_bph[plane + (j4 + 1) * BSTRIDE + e] = h;
        lds_bpl[plane + (j4 + 1) * BSTRIDE + e] = l;
        split_bf(v.z * s, h, l);
        lds_bph[plane + (j4 + 2) * BSTRIDE + e] = h;
        lds_bpl[plane + (j4 + 2) * BSTRIDE + e] = l;
        split_bf(v.w * s, h, l);
        lds_bph[plane + (j4 + 3) * BSTRIDE + e] = h;
        lds_bpl[plane + (j4 + 3) * BSTRIDE + e] = l;
      }
    }
    // prefetch next K-chunk of both streams into cache
    if (kb + 64 < HW) {
      __builtin_prefetch(ws1 + (size_t)(tid & 127) * HW + kb + 64 + ((tid >> 7) << 5), 0, 1);
      __builtin_prefetch(fb + (size_t)(kb + 64 + (tid >> 2)) * FEAT + ((tid & 3) << 5), 0, 1);
    }
    __syncthreads();
    // ---- compute: 2 K-blocks of 32 ----
#pragma unroll
    for (int kb2 = 0; kb2 < 2; ++kb2) {
      int k0 = kb2 * 32 + 16 * hl;
      const u16* pa0h = lds_ah + (mt * 32 + l16) * APITCH + k0;
      const u16* pa0l = lds_al + (mt * 32 + l16) * APITCH + k0;
      v16bf a0h = ldfrag(pa0h, pa0h + 8);
      v16bf a0l = ldfrag(pa0l, pa0l + 8);
      v16bf a1h = ldfrag(pa0h + 16 * APITCH, pa0h + 16 * APITCH + 8);
      v16bf a1l = ldfrag(pa0l + 16 * APITCH, pa0l + 16 * APITCH + 8);
#pragma unroll
      for (int t = 0; t < 2; ++t) {
        int c0 = (ng * 2 + t) * 32 + l16;
        int base = (kb2 * 2 + hl) * BPLANE + c0 * BSTRIDE;
        v16bf b0h = ldfrag(lds_bph + base, lds_bph + base + 8);
        v16bf b0l = ldfrag(lds_bpl + base, lds_bpl + base + 8);
        v16bf b1h = ldfrag(lds_bph + base + 16 * BSTRIDE, lds_bph + base + 16 * BSTRIDE + 8);
        v16bf b1l = ldfrag(lds_bpl + base + 16 * BSTRIDE, lds_bpl + base + 16 * BSTRIDE + 8);
        acc[t][0][0] = wmma_bf(a0h, b0h, acc[t][0][0]);
        acc[t][0][0] = wmma_bf(a0h, b0l, acc[t][0][0]);
        acc[t][0][0] = wmma_bf(a0l, b0h, acc[t][0][0]);
        acc[t][0][1] = wmma_bf(a0h, b1h, acc[t][0][1]);
        acc[t][0][1] = wmma_bf(a0h, b1l, acc[t][0][1]);
        acc[t][0][1] = wmma_bf(a0l, b1h, acc[t][0][1]);
        acc[t][1][0] = wmma_bf(a1h, b0h, acc[t][1][0]);
        acc[t][1][0] = wmma_bf(a1h, b0l, acc[t][1][0]);
        acc[t][1][0] = wmma_bf(a1l, b0h, acc[t][1][0]);
        acc[t][1][1] = wmma_bf(a1h, b1h, acc[t][1][1]);
        acc[t][1][1] = wmma_bf(a1h, b1l, acc[t][1][1]);
        acc[t][1][1] = wmma_bf(a1l, b1h, acc[t][1][1]);
      }
    }
  }
  __syncthreads();

  // ---- bias + relu -> lds_hs[m][c] (fp32, unions the staging buffers) ----
#pragma unroll
  for (int t = 0; t < 2; ++t) {
#pragma unroll
    for (int ns = 0; ns < 2; ++ns) {
      int c = (ng * 2 + t) * 32 + ns * 16 + l16;
#pragma unroll
      for (int ms = 0; ms < 2; ++ms) {
        int m0 = mt * 32 + ms * 16 + 8 * hl;
        v8f v = acc[t][ms][ns];
#pragma unroll
        for (int e = 0; e < 8; ++e) {
          float hv = v[e] + lds_small[m0 + e];
          lds_hs[(m0 + e) * 132 + c] = hv > 0.0f ? hv : 0.0f;
        }
      }
    }
  }
  __syncthreads();

  // ---- token GEMM (8x128 @ 128xC) + bias + emb_pos ----
  {
    int c = tid & 127, tg = tid >> 7;
    float a0 = 0.f, a1 = 0.f, a2 = 0.f, a3 = 0.f;
    const float* w2 = lds_small + 128 + tg * 4 * 128;
    for (int m = 0; m < 128; ++m) {
      float hv = lds_hs[m * 132 + c];
      a0 += w2[m] * hv;
      a1 += w2[128 + m] * hv;
      a2 += w2[256 + m] * hv;
      a3 += w2[384 + m] * hv;
    }
    int cg = cbase + c;
    float* o = outs + (size_t)b * NTOK * FEAT + tg * 4 * FEAT + cg;
    const float* ep = emb_pos + tg * 4 * FEAT + cg;
    const float* b2 = lds_small + 128 + NTOK * 128 + tg * 4;
    o[0]        = a0 + b2[0] + ep[0];
    o[FEAT]     = a1 + b2[1] + ep[FEAT];
    o[2 * FEAT] = a2 + b2[2] + ep[2 * FEAT];
    o[3 * FEAT] = a3 + b2[3] + ep[3 * FEAT];
  }
}

// ---------------------------------------------------------------------------
extern "C" void kernel_launch(void* const* d_in, const int* in_sizes, int n_in,
                              void* d_out, int out_size, void* d_ws, size_t ws_size,
                              hipStream_t stream) {
  const float* feat  = (const float*)d_in[0];
  const float* proto = (const float*)d_in[1];
  // d_in[2] cls_ids, d_in[3] num_classes: ncls==1 & cls_ids==1 -> identity select
  const float* w_d1  = (const float*)d_in[4];
  const float* b_d1  = (const float*)d_in[5];
  const float* w_d2  = (const float*)d_in[6];
  const float* b_d2  = (const float*)d_in[7];
  const float* w_s1  = (const float*)d_in[8];
  const float* b_s1  = (const float*)d_in[9];
  const float* w_s2  = (const float*)d_in[10];
  const float* b_s2  = (const float*)d_in[11];
  // d_in[12] emb_neg unused (one_hot == 1)
  const float* emb_p = (const float*)d_in[13];

  float* out = (float*)d_out;
  char*  ws  = (char*)d_ws;
  float* sim  = (float*)ws;                    // 131072 f32  (512 KB)
  u16*   w1ph = (u16*)(ws + 524288);           // 32768 u16 each (64 KB each)
  u16*   w1pl = (u16*)(ws + 589824);
  u16*   w2ph = (u16*)(ws + 655360);
  u16*   w2pl = (u16*)(ws + 720896);           // total ws: 768 KB

  pack_kernel<<<256, 256, 0, stream>>>(w_d1, w_d2, w1ph, w1pl, w2ph, w2pl);
  sim_kernel<<<16384, 256, 0, stream>>>(feat, proto, sim);
  dense_kernel<<<dim3(64, 32), 256, 0, stream>>>(feat, sim, w1ph, w1pl, w2ph, w2pl,
                                                 b_d1, b_d2, out);
  sparse_kernel<<<dim3(2, 32), 256, 0, stream>>>(feat, sim, w_s1, b_s1, w_s2, b_s2,
                                                 emb_p, out + (size_t)32 * FEAT * HW);
}
